// GCNN_40716289966784
// MI455X (gfx1250) — matrix operand order repaired and checked by reading, not compiled, not run on hardware
//
#include <hip/hip_runtime.h>
#include <hip/hip_bf16.h>

#define NN 40000
#define NE 320000
#define NG 8
#define SENTINEL (-2147483647 - 1)

typedef __attribute__((ext_vector_type(16))) __bf16 v16bf;
typedef __attribute__((ext_vector_type(8)))  __bf16 v8bf;
typedef __attribute__((ext_vector_type(8)))  float  v8f;
typedef unsigned int u32x4 __attribute__((ext_vector_type(4)));
typedef int i32x8 __attribute__((ext_vector_type(8)));
typedef int i32x4 __attribute__((ext_vector_type(4)));

static __device__ __forceinline__ v8f wmma_bf16(v16bf a, v16bf b, v8f c) {
  return __builtin_amdgcn_wmma_f32_16x16x32_bf16(false, a, false, b, (short)0, c, false, false);
}

// A-fragment (16x32 bf16, wave32): lane L -> row L%16; elem e -> K =
// (L/16)*16 + e. Two contiguous 16B LDS reads.
static __device__ __forceinline__ v16bf lds_frag(const __bf16* s, int stride, int row,
                                                 int half, int k0) {
  v8bf lo = *(const v8bf*)(s + row * stride + k0 + half * 16);
  v8bf hi = *(const v8bf*)(s + row * stride + k0 + half * 16 + 8);
  return __builtin_shufflevector(lo, hi, 0,1,2,3,4,5,6,7,8,9,10,11,12,13,14,15);
}

static __device__ __forceinline__ int enc_f32(float f) {
  int k = __float_as_int(f);
  return k >= 0 ? k : (k ^ 0x7fffffff);
}
static __device__ __forceinline__ float dec_f32(int k) {
  int b = k >= 0 ? k : (k ^ 0x7fffffff);
  return __int_as_float(b);
}

// TDM: load a [rows x kcn] bf16 tile (row stride Kpad elems in memory, rows stored
// contiguously kcn elems apart in LDS). D# per CDNA5 ISA.
// This toolchain's builtin is the 6-arg form: (u32x4, i32x8, i32x4, i32x4, i32x8, cpol).
static __device__ __forceinline__ void tdm_load_2d(__bf16* ldsDst, const __bf16* gsrc,
                                                   int Kpad, int tensorRows, int kcn,
                                                   int tileRows) {
  unsigned long long ga = (unsigned long long)(size_t)gsrc;
  u32x4 g0;
  g0[0] = 1u;                                   // count=1, user mode, no gather
  g0[1] = (unsigned)(size_t)ldsDst;             // lds_addr (low 32 bits of flat addr)
  g0[2] = (unsigned)ga;                         // global_addr[31:0]
  g0[3] = (unsigned)((ga >> 32) & 0x01ffffffu) | (2u << 30);  // addr[56:32] | type=2
  i32x8 g1;
  g1[0] = (int)(1u << 16);                      // data_size=1 (2 bytes)
  g1[1] = (int)(((unsigned)Kpad & 0xffffu) << 16);            // tensor_dim0[15:0]
  g1[2] = (int)((((unsigned)Kpad >> 16) & 0xffffu) |
                (((unsigned)tensorRows & 0xffffu) << 16));    // dim0 hi | dim1 lo
  g1[3] = (int)((((unsigned)tensorRows >> 16) & 0xffffu) |
                (((unsigned)kcn & 0xffffu) << 16));           // dim1 hi | tile_dim0
  g1[4] = (int)((unsigned)tileRows & 0xffffu);                // tile_dim1, tile_dim2=0
  g1[5] = Kpad;                                 // tensor_dim0_stride[31:0]
  g1[6] = 0;
  g1[7] = 0;
  i32x4 gz4 = {0, 0, 0, 0};
  i32x8 gz8 = {0, 0, 0, 0, 0, 0, 0, 0};
  __builtin_amdgcn_tensor_load_to_lds(g0, g1, gz4, gz4, gz8, 0);
}

// ---------------- utility kernels ----------------
__global__ void k_zero4(float4* p, size_t n4) {
  size_t i = (size_t)blockIdx.x * 256 + threadIdx.x;
  if (i < n4) p[i] = make_float4(0.f, 0.f, 0.f, 0.f);
}
__global__ void k_fill_i(int* p, int v, size_t n) {
  size_t i = (size_t)blockIdx.x * 256 + threadIdx.x;
  if (i < n) p[i] = v;
}
// W[K x Nout] f32 (python row-major) -> WT[Nout x Kpad] bf16 (zero pad)
__global__ void k_wconv(const float* __restrict__ W, __bf16* __restrict__ WT,
                        int K, int Nout, int Kpad) {
  int i = blockIdx.x * 256 + threadIdx.x;
  if (i < Nout * Kpad) {
    int n = i / Kpad, k = i % Kpad;
    float v = (k < K) ? W[(size_t)k * Nout + n] : 0.0f;
    WT[i] = (__bf16)v;
  }
}

// ---------------- GraphNorm ----------------
__global__ void k_gn_zero(float* st) { if (threadIdx.x < 64) st[threadIdx.x] = 0.0f; }
__global__ void k_gn_sum(const float* __restrict__ x, const int* __restrict__ batch,
                         float* st, int N) {
  int n = blockIdx.x * 256 + threadIdx.x;
  if (n < N) {
    int b = batch[n];
    atomicAdd(&st[b * 3 + 0], x[n * 3 + 0]);
    atomicAdd(&st[b * 3 + 1], x[n * 3 + 1]);
    atomicAdd(&st[b * 3 + 2], x[n * 3 + 2]);
    atomicAdd(&st[24 + b], 1.0f);
  }
}
__global__ void k_gn_mean(float* st) {
  int i = threadIdx.x;
  if (i < 24) { float c = fmaxf(st[24 + i / 3], 1.0f); st[i] /= c; }
}
__global__ void k_gn_center(const float* __restrict__ x, const int* __restrict__ batch,
                            const float* __restrict__ ms, float* st, float* y0, int N) {
  int n = blockIdx.x * 256 + threadIdx.x;
  if (n < N) {
    int b = batch[n];
    for (int f = 0; f < 3; f++) {
      float o = x[n * 3 + f] - st[b * 3 + f] * ms[f];
      y0[n * 3 + f] = o;
      atomicAdd(&st[32 + b * 3 + f], o * o);
    }
  }
}
__global__ void k_gn_var(float* st) {
  int i = threadIdx.x;
  if (i < 24) { float c = fmaxf(st[24 + i / 3], 1.0f); st[32 + i] /= c; }
}
__global__ void k_gn_norm(const int* __restrict__ batch, const float* __restrict__ w,
                          const float* __restrict__ bias, float* y0,
                          const float* __restrict__ st, int N) {
  int n = blockIdx.x * 256 + threadIdx.x;
  if (n < N) {
    int b = batch[n];
    for (int f = 0; f < 3; f++)
      y0[n * 3 + f] = w[f] * y0[n * 3 + f] * rsqrtf(st[32 + b * 3 + f] + 1e-5f) + bias[f];
  }
}

// ---------------- EdgeConv ----------------
// Block: 32 edges x full dout; each wave holds a 16-col tile with 2 row sub-tiles
// so each weight B-fragment feeds 2 WMMAs.
__global__ void k_ec_edge(const float* __restrict__ xin, const int* __restrict__ src,
                          const int* __restrict__ dst,
                          const __bf16* __restrict__ W1T, const float* __restrict__ b1,
                          const __bf16* __restrict__ W2T, const float* __restrict__ b2,
                          int* __restrict__ aggEnc, int E, int din, int K1pad,
                          int dout, int K2pad) {
  __shared__ __align__(16) __bf16 sF[32 * 264];  // K1pad <= 256
  __shared__ __align__(16) __bf16 sH[32 * 264];  // K2pad <= 256
  const int K1p = K1pad + 8, K2p = K2pad + 8;
  const int e0 = blockIdx.x * 32;
  for (int i = threadIdx.x; i < 32 * K1pad; i += 256) {
    int r = i / K1pad, k = i % K1pad;
    float v = 0.0f;
    int e = e0 + r;
    if (e < E && k < 2 * din) {
      int d = dst[e];
      if (k < din) v = xin[(size_t)d * din + k];
      else {
        int s2 = src[e];
        int kk = k - din;
        v = xin[(size_t)s2 * din + kk] - xin[(size_t)d * din + kk];
      }
    }
    sF[r * K1p + k] = (__bf16)v;
  }
  for (int i = threadIdx.x; i < 32 * K2pad; i += 256) {
    int r = i / K2pad, k = i % K2pad;
    sH[r * K2p + k] = (__bf16)0.0f;
  }
  __syncthreads();
  const int wave = threadIdx.x >> 5, lane = threadIdx.x & 31;
  const int half = lane >> 4, lm = lane & 15;
  const int nct = (dout + 15) >> 4;
  // stage 1: h1 = tanh(F @ W1 + b1) -> sH (bf16)
  for (int ct = wave; ct < nct; ct += 8) {
    int n0 = ct * 16;
    v8f acc0 = {}, acc1 = {};
    int nb = min(n0 + lm, dout - 1);
    for (int k0 = 0; k0 < K1pad; k0 += 32) {
      v16bf b = *(const v16bf*)(W1T + (size_t)nb * K1pad + k0 + half * 16);
      v16bf a0 = lds_frag(sF, K1p, lm, half, k0);
      v16bf a1 = lds_frag(sF + 16 * K1p, K1p, lm, half, k0);
      acc0 = wmma_bf16(a0, b, acc0);
      acc1 = wmma_bf16(a1, b, acc1);
    }
    int n = n0 + lm;
    if (n < dout) {
      float bias = b1[n];
      for (int v = 0; v < 8; v++) {
        int r = half * 8 + v;
        sH[r * K2p + n] = (__bf16)tanhf(acc0[v] + bias);
        sH[(16 + r) * K2p + n] = (__bf16)tanhf(acc1[v] + bias);
      }
    }
  }
  __syncthreads();
  // stage 2: h2 = h1 @ W2 + b2 -> encoded atomicMax scatter at dst
  for (int ct = wave; ct < nct; ct += 8) {
    int n0 = ct * 16;
    v8f acc0 = {}, acc1 = {};
    int nb = min(n0 + lm, dout - 1);
    for (int k0 = 0; k0 < K2pad; k0 += 32) {
      v16bf b = *(const v16bf*)(W2T + (size_t)nb * K2pad + k0 + half * 16);
      v16bf a0 = lds_frag(sH, K2p, lm, half, k0);
      v16bf a1 = lds_frag(sH + 16 * K2p, K2p, lm, half, k0);
      acc0 = wmma_bf16(a0, b, acc0);
      acc1 = wmma_bf16(a1, b, acc1);
    }
    int n = n0 + lm;
    if (n < dout) {
      float bias = b2[n];
      for (int v = 0; v < 8; v++) {
        int r = half * 8 + v;
        int e = e0 + r;
        if (e < E)
          atomicMax(&aggEnc[(size_t)dst[e] * dout + n], enc_f32(acc0[v] + bias));
        e = e0 + 16 + r;
        if (e < E)
          atomicMax(&aggEnc[(size_t)dst[e] * dout + n], enc_f32(acc1[v] + bias));
      }
    }
  }
}
__global__ void k_ec_finish(const int* __restrict__ enc, float* __restrict__ y, size_t n) {
  size_t i = (size_t)blockIdx.x * 256 + threadIdx.x;
  if (i < n) {
    int k = enc[i];
    float v = 0.0f;
    if (k != SENTINEL) {
      v = dec_f32(k);
      if (!isfinite(v)) v = 0.0f;
    }
    y[i] = tanhf(v);
  }
}

// ---------------- GraphConv ----------------
__global__ void k_gc_scatter4(const float* __restrict__ x, const int* __restrict__ src,
                              const int* __restrict__ dst, float* __restrict__ agg,
                              int E, int K) {
  size_t i = (size_t)blockIdx.x * 256 + threadIdx.x;
  int Kq = K >> 2;
  size_t total = (size_t)E * Kq;
  if (i < total) {
    int e = (int)(i / Kq), q = (int)(i % Kq);
    const float4 v = *(const float4*)(x + (size_t)src[e] * K + q * 4);
    float* a = agg + (size_t)dst[e] * K + q * 4;
    atomicAdd(a + 0, v.x);
    atomicAdd(a + 1, v.y);
    atomicAdd(a + 2, v.z);
    atomicAdd(a + 3, v.w);
  }
}

// out = tanh(agg@Wr + x@Wl + br) (+residual).  Block: 32 rows x 128 cols, 8 waves.
// Weight tiles staged into LDS by the Tensor Data Mover; activations converted
// f32->bf16 cooperatively into LDS.
#define GKC 64
__global__ void k_gc_gemm(const float* __restrict__ agg, const float* __restrict__ xin,
                          const __bf16* __restrict__ WrT, const __bf16* __restrict__ WlT,
                          const float* __restrict__ br, const float* __restrict__ residual,
                          float* __restrict__ out, int N, int K, int Kpad, int Nout,
                          int applyTanh) {
  __shared__ __align__(16) __bf16 sA[32 * (GKC + 8)];
  __shared__ __align__(16) __bf16 sX[32 * (GKC + 8)];
  __shared__ __align__(16) __bf16 sBr[128 * GKC];
  __shared__ __align__(16) __bf16 sBl[128 * GKC];
  const int Kp = GKC + 8;
  const int m0 = blockIdx.x * 32;
  const int nblk = blockIdx.y * 128;
  const int wave = threadIdx.x >> 5, lane = threadIdx.x & 31;
  const int half = lane >> 4, lm = lane & 15;
  const int nloc = wave * 16 + lm;  // 0..127
  int rows = Nout - nblk;
  if (rows > 128) rows = 128;
  v8f acc0 = {}, acc1 = {};
  for (int kc = 0; kc < Kpad; kc += GKC) {
    // TDM: async-stage both weight tiles (wave0 issues, TENSORcnt waited below)
    if (wave == 0) {
      tdm_load_2d(sBr, WrT + (size_t)nblk * Kpad + kc, Kpad, Nout, GKC, rows);
      tdm_load_2d(sBl, WlT + (size_t)nblk * Kpad + kc, Kpad, Nout, GKC, rows);
    }
    // stage activations (f32 -> bf16)
    for (int i = threadIdx.x; i < 32 * GKC; i += 256) {
      int r = i / GKC, k = i % GKC, gk = kc + k;
      float a = 0.0f, b = 0.0f;
      int m = m0 + r;
      if (m < N && gk < K) {
        a = agg[(size_t)m * K + gk];
        b = xin[(size_t)m * K + gk];
      }
      sA[r * Kp + k] = (__bf16)a;
      sX[r * Kp + k] = (__bf16)b;
    }
    if (wave == 0) __builtin_amdgcn_s_wait_tensorcnt(0);
    __syncthreads();
    if (nblk + wave * 16 < Nout) {
      for (int k0 = 0; k0 < GKC; k0 += 32) {
        // B-fragment: lane holds col nloc, K = k0 + half*16 + e (contiguous 32B)
        v16bf b1 = *(const v16bf*)(sBr + (size_t)nloc * GKC + k0 + half * 16);
        v16bf b2 = *(const v16bf*)(sBl + (size_t)nloc * GKC + k0 + half * 16);
        v16bf a0 = lds_frag(sA, Kp, lm, half, k0);
        v16bf a1 = lds_frag(sA + 16 * Kp, Kp, lm, half, k0);
        v16bf x0 = lds_frag(sX, Kp, lm, half, k0);
        v16bf x1 = lds_frag(sX + 16 * Kp, Kp, lm, half, k0);
        acc0 = wmma_bf16(a0, b1, acc0);
        acc0 = wmma_bf16(x0, b2, acc0);
        acc1 = wmma_bf16(a1, b1, acc1);
        acc1 = wmma_bf16(x1, b2, acc1);
      }
    }
    __syncthreads();
  }
  int n = nblk + nloc;
  if (n < Nout) {
    float bias = br[n];
    for (int v = 0; v < 8; v++) {
      int m = m0 + half * 8 + v;
      if (m < N) {
        float val = acc0[v] + bias;
        if (applyTanh) val = tanhf(val);
        if (residual) val += residual[(size_t)m * Nout + n];
        out[(size_t)m * Nout + n] = val;
      }
      m = m0 + 16 + half * 8 + v;
      if (m < N) {
        float val = acc1[v] + bias;
        if (applyTanh) val = tanhf(val);
        if (residual) val += residual[(size_t)m * Nout + n];
        out[(size_t)m * Nout + n] = val;
      }
    }
  }
}

// final layer 28 -> 2 (no tanh), scalar per node
__global__ void k_gc_last(const float* __restrict__ agg, const float* __restrict__ xin,
                          const float* __restrict__ Wr, const float* __restrict__ br,
                          const float* __restrict__ Wl, float* __restrict__ out, int N) {
  int n = blockIdx.x * 256 + threadIdx.x;
  if (n < N) {
    float o0 = br[0], o1 = br[1];
    for (int k = 0; k < 28; k++) {
      float a = agg[(size_t)n * 28 + k], xx = xin[(size_t)n * 28 + k];
      o0 += a * Wr[k * 2 + 0] + xx * Wl[k * 2 + 0];
      o1 += a * Wr[k * 2 + 1] + xx * Wl[k * 2 + 1];
    }
    out[n * 2 + 0] = o0;
    out[n * 2 + 1] = o1;
  }
}

// ---------------- host ----------------
static inline unsigned cdiv(size_t a, size_t b) { return (unsigned)((a + b - 1) / b); }
static inline int rup32(int k) { return (k + 31) & ~31; }

extern "C" void kernel_launch(void* const* d_in, const int* in_sizes, int n_in,
                              void* d_out, int out_size, void* d_ws, size_t ws_size,
                              hipStream_t stream) {
  const float *x, *nw, *nb, *nms;
  const int *ei, *batch;
  const float *ecW1[4], *ecb1[4], *ecW2[4], *ecb2[4];
  const float *gcWr[10], *gcbr[10], *gcWl[10];
  if (in_sizes[0] == 3 * NN) {  // insertion order
    x = (const float*)d_in[0];
    ei = (const int*)d_in[1];
    batch = (const int*)d_in[2];
    nw = (const float*)d_in[3]; nb = (const float*)d_in[4]; nms = (const float*)d_in[5];
    for (int l = 0; l < 4; l++) {
      ecW1[l] = (const float*)d_in[6 + 4 * l];
      ecb1[l] = (const float*)d_in[7 + 4 * l];
      ecW2[l] = (const float*)d_in[8 + 4 * l];
      ecb2[l] = (const float*)d_in[9 + 4 * l];
    }
    for (int l = 0; l < 10; l++) {
      gcWr[l] = (const float*)d_in[22 + 3 * l];
      gcbr[l] = (const float*)d_in[23 + 3 * l];
      gcWl[l] = (const float*)d_in[24 + 3 * l];
    }
  } else {  // jax pytree sorted-key order
    batch = (const int*)d_in[0];
    ei = (const int*)d_in[1];
    for (int l = 0; l < 4; l++) {
      ecW1[l] = (const float*)d_in[2 + 4 * l];
      ecW2[l] = (const float*)d_in[3 + 4 * l];
      ecb1[l] = (const float*)d_in[4 + 4 * l];
      ecb2[l] = (const float*)d_in[5 + 4 * l];
    }
    for (int l = 0; l < 10; l++) {
      gcWl[l] = (const float*)d_in[18 + 3 * l];
      gcWr[l] = (const float*)d_in[19 + 3 * l];
      gcbr[l] = (const float*)d_in[20 + 3 * l];
    }
    nb = (const float*)d_in[48]; nms = (const float*)d_in[49]; nw = (const float*)d_in[50];
    x = (const float*)d_in[51];
  }
  const int N = NN, E = NE;
  const int* src = ei;
  const int* dst = ei + E;

  static const int ecDin[4] = {3, 28, 64, 128}, ecDout[4] = {28, 64, 128, 256};
  static const int gcDin[10] = {256, 256, 512, 1024, 1024, 512, 256, 128, 64, 28};
  static const int gcDout[10] = {256, 512, 1024, 1024, 512, 256, 128, 64, 28, 2};

  char* base = (char*)d_ws;
  size_t cur = 0;
  auto alloc = [&](size_t bytes) {
    size_t o = cur;
    cur = (cur + bytes + 255) & ~(size_t)255;
    return o;
  };
  __bf16 *ecW1T[4], *ecW2T[4], *gcWrT[10], *gcWlT[10];
  for (int l = 0; l < 4; l++) {
    int K1p = rup32(2 * ecDin[l]), K2p = rup32(ecDout[l]);
    ecW1T[l] = (__bf16*)(base + alloc((size_t)ecDout[l] * K1p * 2));
    ecW2T[l] = (__bf16*)(base + alloc((size_t)ecDout[l] * K2p * 2));
  }
  for (int l = 0; l < 9; l++) {
    int Kp = rup32(gcDin[l]);
    gcWrT[l] = (__bf16*)(base + alloc((size_t)gcDout[l] * Kp * 2));
    gcWlT[l] = (__bf16*)(base + alloc((size_t)gcDout[l] * Kp * 2));
  }
  float* st = (float*)(base + alloc(64 * 4));
  float* y0 = (float*)(base + alloc((size_t)N * 3 * 4));
  float* y1 = (float*)(base + alloc((size_t)N * 28 * 4));
  float* y2 = (float*)(base + alloc((size_t)N * 64 * 4));
  float* y3 = (float*)(base + alloc((size_t)N * 128 * 4));
  float* y5 = (float*)(base + alloc((size_t)N * 256 * 4));
  float* y6 = (float*)(base + alloc((size_t)N * 512 * 4));
  float* y7 = (float*)(base + alloc((size_t)N * 1024 * 4));
  float* bufA = (float*)(base + alloc((size_t)N * 1024 * 4));
  float* bufB = (float*)(base + alloc((size_t)N * 1024 * 4));
  float* aggF = (float*)(base + alloc((size_t)N * 1024 * 4));
  int* aggEnc = (int*)aggF;

  // weight conversion (bf16, transposed, K padded to 32)
  for (int l = 0; l < 4; l++) {
    int K1 = 2 * ecDin[l], K1p = rup32(K1), K2 = ecDout[l], K2p = rup32(K2), D = ecDout[l];
    k_wconv<<<cdiv((size_t)D * K1p, 256), 256, 0, stream>>>(ecW1[l], ecW1T[l], K1, D, K1p);
    k_wconv<<<cdiv((size_t)D * K2p, 256), 256, 0, stream>>>(ecW2[l], ecW2T[l], K2, D, K2p);
  }
  for (int l = 0; l < 9; l++) {
    int K = gcDin[l], Kp = rup32(K), D = gcDout[l];
    k_wconv<<<cdiv((size_t)D * Kp, 256), 256, 0, stream>>>(gcWr[l], gcWrT[l], K, D, Kp);
    k_wconv<<<cdiv((size_t)D * Kp, 256), 256, 0, stream>>>(gcWl[l], gcWlT[l], K, D, Kp);
  }

  // GraphNorm -> y0
  k_gn_zero<<<1, 64, 0, stream>>>(st);
  k_gn_sum<<<cdiv(N, 256), 256, 0, stream>>>(x, batch, st, N);
  k_gn_mean<<<1, 64, 0, stream>>>(st);
  k_gn_center<<<cdiv(N, 256), 256, 0, stream>>>(x, batch, nms, st, y0, N);
  k_gn_var<<<1, 64, 0, stream>>>(st);
  k_gn_norm<<<cdiv(N, 256), 256, 0, stream>>>(batch, nw, nb, y0, st, N);

  // EdgeConv chain: y0->y1->y2->y3->bufA(=y4)
  const float* ecin[4] = {y0, y1, y2, y3};
  float* ecout[4] = {y1, y2, y3, bufA};
  for (int l = 0; l < 4; l++) {
    int din = ecDin[l], dout = ecDout[l];
    int K1p = rup32(2 * din), K2p = rup32(dout);
    size_t tot = (size_t)N * dout;
    k_fill_i<<<cdiv(tot, 256), 256, 0, stream>>>(aggEnc, SENTINEL, tot);
    k_ec_edge<<<cdiv(E, 32), 256, 0, stream>>>(ecin[l], src, dst, ecW1T[l], ecb1[l],
                                               ecW2T[l], ecb2[l], aggEnc, E, din, K1p,
                                               dout, K2p);
    k_ec_finish<<<cdiv(tot, 256), 256, 0, stream>>>(aggEnc, ecout[l], tot);
  }

  // GraphConv chain with residuals
  const float* gin[9] = {bufA, y5, y6, y7, bufA, bufB, bufA, bufB, bufA};
  float* gout[9] = {y5, y6, y7, bufA, bufB, bufA, bufB, bufA, bufB};
  const float* gres[9] = {nullptr, nullptr, nullptr, y7, y6, y5, y3, y2, y1};
  for (int l = 0; l < 9; l++) {
    int K = gcDin[l], Kp = rup32(K), D = gcDout[l];
    size_t aggTot = (size_t)N * K;
    k_zero4<<<cdiv(aggTot / 4, 256), 256, 0, stream>>>((float4*)aggF, aggTot / 4);
    k_gc_scatter4<<<cdiv((size_t)E * (K / 4), 256), 256, 0, stream>>>(gin[l], src, dst,
                                                                      aggF, E, K);
    dim3 g(cdiv(N, 32), cdiv(D, 128));
    k_gc_gemm<<<g, 256, 0, stream>>>(aggF, gin[l], gcWrT[l], gcWlT[l], gcbr[l], gres[l],
                                     gout[l], N, K, Kp, D, 1);
  }
  // final: gc[9] 28->2, no tanh, output to d_out
  {
    size_t aggTot = (size_t)N * 28;
    k_zero4<<<cdiv(aggTot / 4, 256), 256, 0, stream>>>((float4*)aggF, aggTot / 4);
    k_gc_scatter4<<<cdiv((size_t)E * 7, 256), 256, 0, stream>>>(bufB, src, dst, aggF, E, 28);
    k_gc_last<<<cdiv(N, 256), 256, 0, stream>>>(aggF, bufB, gcWr[9], gcbr[9], gcWl[9],
                                                (float*)d_out, N);
  }
  (void)n_in; (void)out_size; (void)ws_size; (void)in_sizes;
}
